// GraphEncoder_46849503265250
// MI455X (gfx1250) — matrix operand order repaired and checked by reading, not compiled, not run on hardware
//
#include <hip/hip_runtime.h>
#include <hip/hip_bf16.h>

// ---------------- problem constants ----------------
#define NN   50000
#define TT   24
#define DIN  16
#define HHID 64
#define SDIM 9
#define EE   1600000
#define NG   500

typedef __attribute__((ext_vector_type(16))) __bf16 v16bf;
typedef __attribute__((ext_vector_type(8)))  float  v8f;

// float -> bf16 round-to-nearest-even
__device__ __forceinline__ __bf16 f2bf(float f) {
    unsigned u = __builtin_bit_cast(unsigned, f);
    unsigned r = u + 0x7FFFu + ((u >> 16) & 1u);
    unsigned short s = (unsigned short)(r >> 16);
    return __builtin_bit_cast(__bf16, s);
}

// A-fragment (16-bit, 16x32) K index for element e, lane-half h
__device__ __forceinline__ int amapK(int e, int half) {
    return e + 16 * half;
}

__device__ __forceinline__ v8f wmma_bf16(v16bf a, v16bf b, v8f c) {
    return __builtin_amdgcn_wmma_f32_16x16x32_bf16(
        /*neg_a=*/false, a, /*neg_b=*/false, b,
        /*c_mod=*/(short)0, c, /*reuse_a=*/false, /*reuse_b=*/false);
}

__device__ __forceinline__ float sigm(float x)  { return 1.f / (1.f + __expf(-x)); }
__device__ __forceinline__ float tanhfast(float x) {
    float e = __expf(-2.f * x);
    return (1.f - e) / (1.f + e);
}

// =====================================================================
// Kernel 1: fused LSTM (final hidden) + l1 linear. Wave = 16 nodes.
// Gate columns are processed one 16-wide group at a time so only the
// 4 C-fragments (i,f,g,o) for that group are live -> ~150 VGPRs total,
// no s_set_vgpr_msb traffic, higher occupancy.
// =====================================================================
__global__ __launch_bounds__(256) void lstm_l1_kernel(
    const float* __restrict__ x,    const float* __restrict__ sdi,
    const float* __restrict__ W_ih, const float* __restrict__ W_hh,
    const float* __restrict__ b_ih, const float* __restrict__ b_hh,
    const float* __restrict__ l1W,  const float* __restrict__ l1b,
    float* __restrict__ z)
{
    // B-fragments in LDS: [tile][lane][elem], elem K = e + 16*(lane>>4)
    __shared__ __align__(32) __bf16 sBx[16][32][16];     // W_ih, K 0..15 valid   (16 KB)
    __shared__ __align__(32) __bf16 sBh[2][16][32][16];  // W_hh, two K-chunks    (32 KB)
    __shared__ __align__(32) __bf16 sB1[3][4][32][16];   // l1_W, 3 K-chunks      (12 KB)
    __shared__ __align__(32) __bf16 sH[8][16][64];       // per-wave h staging    (16 KB)

    const int tid = threadIdx.x;

    for (int idx = tid; idx < 16 * 32 * 16; idx += 256) {
        int e = idx & 15, ln = (idx >> 4) & 31, j = idx >> 9;
        int K = e + 16 * (ln >> 4);
        int col = (ln & 15) + 16 * j;
        sBx[j][ln][e] = f2bf(K < DIN ? W_ih[col * DIN + K] : 0.f);
    }
    for (int idx = tid; idx < 2 * 16 * 32 * 16; idx += 256) {
        int e = idx & 15, ln = (idx >> 4) & 31, j = (idx >> 9) & 15, c = idx >> 13;
        int K = e + 16 * (ln >> 4) + 32 * c;
        int col = (ln & 15) + 16 * j;
        sBh[c][j][ln][e] = f2bf(W_hh[col * HHID + K]);
    }
    for (int idx = tid; idx < 3 * 4 * 32 * 16; idx += 256) {
        int e = idx & 15, ln = (idx >> 4) & 31, j = (idx >> 9) & 3, c = idx >> 11;
        int K = e + 16 * (ln >> 4);
        int col = (ln & 15) + 16 * j;
        float w;
        if (c < 2) w = l1W[col * (HHID + SDIM) + K + 32 * c];
        else       w = (K < SDIM) ? l1W[col * (HHID + SDIM) + HHID + K] : 0.f;
        sB1[c][j][ln][e] = f2bf(w);
    }
    __syncthreads();

    const int wave = tid >> 5;
    const int lane = tid & 31;
    const int half = lane >> 4;
    const int ml   = lane & 15;
    const int gw   = blockIdx.x * 8 + wave;
    if (gw >= NN / 16) return;
    const int base = gw * 16;

    float bsum[16];
    #pragma unroll
    for (int j = 0; j < 16; ++j) bsum[j] = b_ih[ml + 16 * j] + b_hh[ml + 16 * j];

    v16bf aH0, aH1;
    #pragma unroll
    for (int e = 0; e < 16; ++e) { aH0[e] = f2bf(0.f); aH1[e] = f2bf(0.f); }

    v8f cst[4];
    #pragma unroll
    for (int j = 0; j < 4; ++j)
        #pragma unroll
        for (int r = 0; r < 8; ++r) cst[j][r] = 0.f;

    for (int t = 0; t < TT; ++t) {
        // x_t A-fragment (K<16 valid)
        v16bf aX;
        #pragma unroll
        for (int e = 0; e < 16; ++e) {
            int K = amapK(e, half);
            aX[e] = (K < DIN) ? f2bf(x[(size_t)(base + ml) * (TT * DIN) + t * DIN + K])
                              : f2bf(0.f);
        }

        // process gate column-groups: tiles (jc, jc+4, jc+8, jc+12) = (i,f,g,o)
        #pragma unroll
        for (int jc = 0; jc < 4; ++jc) {
            v8f gi, gf, gg, go;
            #pragma unroll
            for (int r = 0; r < 8; ++r) {
                gi[r] = bsum[jc];
                gf[r] = bsum[jc + 4];
                gg[r] = bsum[jc + 8];
                go[r] = bsum[jc + 12];
            }
            gi = wmma_bf16(aX,  *(const v16bf*)&sBx[jc][lane][0],         gi);
            gi = wmma_bf16(aH0, *(const v16bf*)&sBh[0][jc][lane][0],      gi);
            gi = wmma_bf16(aH1, *(const v16bf*)&sBh[1][jc][lane][0],      gi);
            gf = wmma_bf16(aX,  *(const v16bf*)&sBx[jc + 4][lane][0],     gf);
            gf = wmma_bf16(aH0, *(const v16bf*)&sBh[0][jc + 4][lane][0],  gf);
            gf = wmma_bf16(aH1, *(const v16bf*)&sBh[1][jc + 4][lane][0],  gf);
            gg = wmma_bf16(aX,  *(const v16bf*)&sBx[jc + 8][lane][0],     gg);
            gg = wmma_bf16(aH0, *(const v16bf*)&sBh[0][jc + 8][lane][0],  gg);
            gg = wmma_bf16(aH1, *(const v16bf*)&sBh[1][jc + 8][lane][0],  gg);
            go = wmma_bf16(aX,  *(const v16bf*)&sBx[jc + 12][lane][0],    go);
            go = wmma_bf16(aH0, *(const v16bf*)&sBh[0][jc + 12][lane][0], go);
            go = wmma_bf16(aH1, *(const v16bf*)&sBh[1][jc + 12][lane][0], go);

            // nonlinearities (torch order i,f,g,o), state update, h -> LDS
            #pragma unroll
            for (int r = 0; r < 8; ++r) {
                float iv = sigm(gi[r]);
                float fv = sigm(gf[r]);
                float gv = tanhfast(gg[r]);
                float ov = sigm(go[r]);
                float cc = fv * cst[jc][r] + iv * gv;
                cst[jc][r] = cc;
                float hv = ov * tanhfast(cc);
                sH[wave][r + 8 * half][ml + 16 * jc] = f2bf(hv);
            }
        }
        // intra-wave cross-lane LDS dependency: program order + DS-counter wait
        asm volatile("s_wait_dscnt 0" ::: "memory");

        // reload h as A-fragments for next step
        #pragma unroll
        for (int e = 0; e < 16; ++e) {
            int K = amapK(e, half);
            aH0[e] = sH[wave][ml][K];
            aH1[e] = sH[wave][ml][K + 32];
        }
    }

    // ---- l1: z = [h | sdi] @ l1_W^T + l1_b ----
    v16bf aS;
    #pragma unroll
    for (int e = 0; e < 16; ++e) {
        int K = amapK(e, half);
        aS[e] = (K < SDIM) ? f2bf(sdi[(size_t)(base + ml) * SDIM + K]) : f2bf(0.f);
    }
    #pragma unroll
    for (int j = 0; j < 4; ++j) {
        v8f zz;
        #pragma unroll
        for (int r = 0; r < 8; ++r) zz[r] = l1b[ml + 16 * j];
        zz = wmma_bf16(aH0, *(const v16bf*)&sB1[0][j][lane][0], zz);
        zz = wmma_bf16(aH1, *(const v16bf*)&sB1[1][j][lane][0], zz);
        zz = wmma_bf16(aS,  *(const v16bf*)&sB1[2][j][lane][0], zz);
        #pragma unroll
        for (int r = 0; r < 8; ++r)
            z[(size_t)(base + r + 8 * half) * HHID + ml + 16 * j] = zz[r];
    }
}

// =====================================================================
// Kernel 2: hW = z @ gcn1_W^T  (no bias; PyG adds bias post-aggregation)
// =====================================================================
__global__ __launch_bounds__(256) void gcn1_gemm_kernel(
    const float* __restrict__ z, const float* __restrict__ W,
    float* __restrict__ hW)
{
    __shared__ __align__(32) __bf16 sB[2][4][32][16];   // 8 KB
    const int tid = threadIdx.x;
    for (int idx = tid; idx < 2 * 4 * 32 * 16; idx += 256) {
        int e = idx & 15, ln = (idx >> 4) & 31, j = (idx >> 9) & 3, c = idx >> 11;
        int K = e + 16 * (ln >> 4) + 32 * c;
        int col = (ln & 15) + 16 * j;
        sB[c][j][ln][e] = f2bf(W[col * HHID + K]);
    }
    __syncthreads();

    const int wave = tid >> 5, lane = tid & 31;
    const int half = lane >> 4, ml = lane & 15;
    const int gw = blockIdx.x * 8 + wave;
    if (gw >= NN / 16) return;
    const int base = gw * 16;

    v16bf a0, a1;
    #pragma unroll
    for (int e = 0; e < 16; ++e) {
        int K = amapK(e, half);
        a0[e] = f2bf(z[(size_t)(base + ml) * HHID + K]);
        a1[e] = f2bf(z[(size_t)(base + ml) * HHID + K + 32]);
    }
    #pragma unroll
    for (int j = 0; j < 4; ++j) {
        v8f acc;
        #pragma unroll
        for (int r = 0; r < 8; ++r) acc[r] = 0.f;
        acc = wmma_bf16(a0, *(const v16bf*)&sB[0][j][lane][0], acc);
        acc = wmma_bf16(a1, *(const v16bf*)&sB[1][j][lane][0], acc);
        #pragma unroll
        for (int r = 0; r < 8; ++r)
            hW[(size_t)(base + r + 8 * half) * HHID + ml + 16 * j] = acc[r];
    }
}

// ---------------- scatter / pooling kernels ----------------
__global__ void init_kernel(float* acc1, float* acc2, float* deg,
                            float* sums, float* cnts)
{
    int i = blockIdx.x * blockDim.x + threadIdx.x;
    if (i < NN * HHID) acc1[i] = 0.f;
    if (i < NN) { acc2[i] = 0.f; deg[i] = 1.f; }   // self-loop contributes 1
    if (i < NG) { sums[i] = 0.f; cnts[i] = 0.f; }
}

__global__ void deg_kernel(const int* __restrict__ ei, float* __restrict__ deg)
{
    int e = blockIdx.x * blockDim.x + threadIdx.x;
    if (e < EE) atomicAdd(&deg[ei[EE + e]], 1.0f);  // in-degree at dst
}

__global__ void rsqrt_kernel(float* deg)
{
    int n = blockIdx.x * blockDim.x + threadIdx.x;
    if (n < NN) deg[n] = rsqrtf(deg[n]);            // deg >= 1 always
}

__global__ void msg1_kernel(const int* __restrict__ ei, const float* __restrict__ hW,
                            const float* __restrict__ dis, float* __restrict__ acc1)
{
    long long gid = (long long)blockIdx.x * blockDim.x + threadIdx.x;
    if (gid >= (long long)EE * HHID) return;
    int e = (int)(gid >> 6);
    int f = (int)(gid & 63);
    int s = ei[e], d = ei[EE + e];
    float nrm = dis[s] * dis[d];
    atomicAdd(&acc1[(size_t)d * HHID + f], hW[(size_t)s * HHID + f] * nrm);
}

__global__ void finalize1_kernel(const float* __restrict__ acc1, const float* __restrict__ hW,
                                 const float* __restrict__ dis, const float* __restrict__ b1,
                                 const float* __restrict__ W2, float* __restrict__ tb)
{
    int n = blockIdx.x * blockDim.x + threadIdx.x;
    if (n >= NN) return;
    float invd = dis[n] * dis[n];                   // self-loop norm = 1/deg
    float acc = 0.f;
    #pragma unroll 4
    for (int f = 0; f < HHID; ++f) {
        float y = acc1[(size_t)n * HHID + f] + hW[(size_t)n * HHID + f] * invd + b1[f];
        y = fmaxf(y, 0.f);                          // ReLU
        acc += y * W2[f];                           // gcn2 matmul (64 -> 1)
    }
    tb[n] = acc;
}

__global__ void msg2_kernel(const int* __restrict__ ei, const float* __restrict__ tb,
                            const float* __restrict__ dis, float* __restrict__ acc2)
{
    int e = blockIdx.x * blockDim.x + threadIdx.x;
    if (e >= EE) return;
    int s = ei[e], d = ei[EE + e];
    atomicAdd(&acc2[d], tb[s] * dis[s] * dis[d]);
}

__global__ void pool_accum_kernel(const float* __restrict__ acc2, const float* __restrict__ tb,
                                  const float* __restrict__ dis, const float* __restrict__ b2,
                                  const int* __restrict__ batch,
                                  float* __restrict__ sums, float* __restrict__ cnts)
{
    int n = blockIdx.x * blockDim.x + threadIdx.x;
    if (n >= NN) return;
    float z2 = acc2[n] + tb[n] * dis[n] * dis[n] + b2[0];
    int g = batch[n];
    atomicAdd(&sums[g], z2);
    atomicAdd(&cnts[g], 1.0f);
}

__global__ void pool_final_kernel(const float* __restrict__ sums,
                                  const float* __restrict__ cnts, float* __restrict__ out)
{
    int g = blockIdx.x * blockDim.x + threadIdx.x;
    if (g < NG) out[g] = sums[g] / fmaxf(cnts[g], 1.0f);
}

// =====================================================================
extern "C" void kernel_launch(void* const* d_in, const int* in_sizes, int n_in,
                              void* d_out, int out_size, void* d_ws, size_t ws_size,
                              hipStream_t stream)
{
    const float* x    = (const float*)d_in[0];
    const float* sdi  = (const float*)d_in[1];
    const int*   ei   = (const int*)  d_in[2];
    const int*   bat  = (const int*)  d_in[3];
    const float* W_ih = (const float*)d_in[4];
    const float* W_hh = (const float*)d_in[5];
    const float* b_ih = (const float*)d_in[6];
    const float* b_hh = (const float*)d_in[7];
    const float* l1W  = (const float*)d_in[8];
    const float* l1b  = (const float*)d_in[9];
    const float* g1W  = (const float*)d_in[10];
    const float* g1b  = (const float*)d_in[11];
    const float* g2W  = (const float*)d_in[12];
    const float* g2b  = (const float*)d_in[13];
    float* out = (float*)d_out;

    float* ws   = (float*)d_ws;
    float* z    = ws;                      // N*64
    float* hW   = ws + (size_t)NN * 64;    // N*64
    float* acc1 = hW + (size_t)NN * 64;    // N*64
    float* tb   = acc1 + (size_t)NN * 64;  // N
    float* acc2 = tb + NN;                 // N
    float* dis  = acc2 + NN;               // N (degree, then rsqrt(deg))
    float* sums = dis + NN;                // 512
    float* cnts = sums + 512;              // 512

    const int waves = NN / 16;             // 3125 (exact)
    const int gemmBlocks = (waves + 7) / 8;

    init_kernel<<<(NN * HHID + 255) / 256, 256, 0, stream>>>(acc1, acc2, dis, sums, cnts);

    lstm_l1_kernel<<<gemmBlocks, 256, 0, stream>>>(x, sdi, W_ih, W_hh, b_ih, b_hh,
                                                   l1W, l1b, z);

    deg_kernel<<<(EE + 255) / 256, 256, 0, stream>>>(ei, dis);
    rsqrt_kernel<<<(NN + 255) / 256, 256, 0, stream>>>(dis);

    gcn1_gemm_kernel<<<gemmBlocks, 256, 0, stream>>>(z, g1W, hW);

    {
        long long tot = (long long)EE * HHID;
        int blocks = (int)((tot + 255) / 256);
        msg1_kernel<<<blocks, 256, 0, stream>>>(ei, hW, dis, acc1);
    }

    finalize1_kernel<<<(NN + 255) / 256, 256, 0, stream>>>(acc1, hW, dis, g1b, g2W, tb);
    msg2_kernel<<<(EE + 255) / 256, 256, 0, stream>>>(ei, tb, dis, acc2);
    pool_accum_kernel<<<(NN + 255) / 256, 256, 0, stream>>>(acc2, tb, dis, g2b, bat,
                                                            sums, cnts);
    pool_final_kernel<<<(NG + 255) / 256, 256, 0, stream>>>(sums, cnts, out);
}
